// GCN_28321014350089
// MI455X (gfx1250) — compile-verified, matrix operand order checked by reading, not run on hardware
//
#include <hip/hip_runtime.h>
#include <hip/hip_bf16.h>

// ---------------------------------------------------------------------------
// 2-layer GCN: pack weights (bf16, transposed) -> gemm1 (WMMA bf16) ->
// edge scatter-add -> gemm2 (WMMA bf16, fused bias+relu on A) ->
// edge scatter-add -> bias + log_softmax.
// ---------------------------------------------------------------------------

typedef __attribute__((ext_vector_type(16))) __bf16 v16bf;
typedef __attribute__((ext_vector_type(8)))  float  v8f;

#define GCN_N        100000
#define GCN_INF      512
#define GCN_HID      128
#define GCN_CLS      40
#define GCN_CLSP     48      // classes padded to 3 x 16 WMMA tiles

// -------------------- weight pre-pack (per launch, deterministic) -----------
// W1t[c][k] = bf16(W1[k][c])   (128 x 512)
// W2t[c][k] = bf16(W2[k][c])   (48 x 128, cols 40..47 zero)
__global__ void __launch_bounds__(256)
gcn_pack_weights(const float* __restrict__ W1, const float* __restrict__ W2,
                 __bf16* __restrict__ W1t, __bf16* __restrict__ W2t) {
    const int i = blockIdx.x * 256 + threadIdx.x;
    if (i < GCN_HID * GCN_INF) {
        const int c = i >> 9;        // 0..127
        const int k = i & 511;       // 0..511
        W1t[i] = (__bf16)W1[(size_t)k * GCN_HID + c];
    }
    const int j = i - GCN_HID * GCN_INF;
    if (j >= 0 && j < GCN_CLSP * GCN_HID) {
        const int c = j >> 7;        // 0..47
        const int k = j & 127;       // 0..127
        W2t[j] = (__bf16)((c < GCN_CLS) ? W2[(size_t)k * GCN_CLS + c] : 0.0f);
    }
}

// -------------------- GEMM1: m = x @ W1  (N x 512) @ (512 x 128) ------------
// Block: 128 threads = 4 waves, 32 output rows per block.
// Wave w: cols [32w, 32w+32), 2 row-tiles x 2 col-tiles = 4 WMMA accs.
#define G1_LDA  516   // 512 + 4 bf16 pad: 8B-aligned rows, bank-conflict-free
#define G1_ROWS 32

__global__ void __launch_bounds__(128)
gcn_gemm1_wmma(const float* __restrict__ x, const __bf16* __restrict__ W1t,
               float* __restrict__ m1) {
    __shared__ __bf16 ldsA[G1_ROWS * G1_LDA];

    const int row0 = blockIdx.x * G1_ROWS;
    const int lane = threadIdx.x & 31;
    const int wv   = threadIdx.x >> 5;

    // Stage A tile (32 x 512 f32 -> bf16) into LDS, coalesced float4 loads.
    for (int i = threadIdx.x; i < G1_ROWS * 128; i += 128) {
        const int r  = i >> 7;       // 0..31
        const int c4 = i & 127;      // float4 index within row
        const float4 v = ((const float4*)(x + (size_t)(row0 + r) * GCN_INF))[c4];
        union { __bf16 h[4]; unsigned long long u; } pk;
        pk.h[0] = (__bf16)v.x; pk.h[1] = (__bf16)v.y;
        pk.h[2] = (__bf16)v.z; pk.h[3] = (__bf16)v.w;
        *(unsigned long long*)&ldsA[r * G1_LDA + c4 * 4] = pk.u;
    }
    __syncthreads();

    const int arow  = lane & 15;            // A row (M) within tile
    const int khalf = (lane >> 4) << 4;     // K sub-half: 0 or 16
    const int col0  = wv * 32;

    v8f acc00 = {}, acc01 = {}, acc10 = {}, acc11 = {};

    // B fragments: 16 contiguous bf16 per lane from transposed weights (32B).
    const __bf16* bb = W1t + (size_t)(col0 + (lane & 15)) * GCN_INF + khalf;

    for (int kb = 0; kb < GCN_INF; kb += 32) {
        v16bf a0, a1;
        {
            const unsigned long long* p0 =
                (const unsigned long long*)&ldsA[arow * G1_LDA + kb + khalf];
            const unsigned long long* p1 =
                (const unsigned long long*)&ldsA[(16 + arow) * G1_LDA + kb + khalf];
            unsigned long long* d0 = (unsigned long long*)&a0;
            unsigned long long* d1 = (unsigned long long*)&a1;
            d0[0] = p0[0]; d0[1] = p0[1]; d0[2] = p0[2]; d0[3] = p0[3];
            d1[0] = p1[0]; d1[1] = p1[1]; d1[2] = p1[2]; d1[3] = p1[3];
        }
        const v16bf b0 = *(const v16bf*)(bb + kb);
        const v16bf b1 = *(const v16bf*)(bb + (size_t)16 * GCN_INF + kb);

        acc00 = __builtin_amdgcn_wmma_f32_16x16x32_bf16(false, a0, false, b0,
                                                        (short)0, acc00, false, false);
        acc01 = __builtin_amdgcn_wmma_f32_16x16x32_bf16(false, a0, false, b1,
                                                        (short)0, acc01, false, false);
        acc10 = __builtin_amdgcn_wmma_f32_16x16x32_bf16(false, a1, false, b0,
                                                        (short)0, acc10, false, false);
        acc11 = __builtin_amdgcn_wmma_f32_16x16x32_bf16(false, a1, false, b1,
                                                        (short)0, acc11, false, false);
    }

    // C layout: lanes 0-15 -> M = r, lanes 16-31 -> M = 8 + r; N = lane & 15.
    const int orow = (lane >> 4) << 3;
    float* op0 = m1 + (size_t)(row0 + orow) * GCN_HID + col0 + (lane & 15);
    float* op1 = m1 + (size_t)(row0 + 16 + orow) * GCN_HID + col0 + (lane & 15);
#pragma unroll
    for (int r = 0; r < 8; ++r) {
        op0[(size_t)r * GCN_HID]      = acc00[r];
        op0[(size_t)r * GCN_HID + 16] = acc01[r];
        op1[(size_t)r * GCN_HID]      = acc10[r];
        op1[(size_t)r * GCN_HID + 16] = acc11[r];
    }
}

// -------- GEMM2: t = relu(agg1 + b1) @ W2   (N x 128) @ (128 x 40) ----------
// Block: 96 threads = 3 waves, 32 rows; cols padded 40 -> 48 (3 tiles of 16).
#define G2_LDA  132   // 128 + 4 pad
#define G2_ROWS 32

__global__ void __launch_bounds__(96)
gcn_gemm2_wmma(const float* __restrict__ agg1, const float* __restrict__ b1,
               const __bf16* __restrict__ W2t, float* __restrict__ t2) {
    __shared__ __bf16 ldsA[G2_ROWS * G2_LDA];

    const int row0 = blockIdx.x * G2_ROWS;
    const int lane = threadIdx.x & 31;
    const int wv   = threadIdx.x >> 5;   // 0..2

    // Stage A tile with fused bias + relu (f32 -> bf16).
    for (int i = threadIdx.x; i < G2_ROWS * GCN_HID; i += 96) {
        const int r = i >> 7;
        const int c = i & 127;
        float v = agg1[(size_t)(row0 + r) * GCN_HID + c] + b1[c];
        v = fmaxf(v, 0.0f);
        ldsA[r * G2_LDA + c] = (__bf16)v;
    }
    __syncthreads();

    const int arow  = lane & 15;
    const int khalf = (lane >> 4) << 4;
    const int col   = wv * 16 + (lane & 15);

    v8f acc0 = {}, acc1 = {};
    const __bf16* bb = W2t + (size_t)col * GCN_HID + khalf;

    for (int kb = 0; kb < GCN_HID; kb += 32) {
        v16bf a0, a1;
        {
            const unsigned long long* p0 =
                (const unsigned long long*)&ldsA[arow * G2_LDA + kb + khalf];
            const unsigned long long* p1 =
                (const unsigned long long*)&ldsA[(16 + arow) * G2_LDA + kb + khalf];
            unsigned long long* d0 = (unsigned long long*)&a0;
            unsigned long long* d1 = (unsigned long long*)&a1;
            d0[0] = p0[0]; d0[1] = p0[1]; d0[2] = p0[2]; d0[3] = p0[3];
            d1[0] = p1[0]; d1[1] = p1[1]; d1[2] = p1[2]; d1[3] = p1[3];
        }
        const v16bf b = *(const v16bf*)(bb + kb);
        acc0 = __builtin_amdgcn_wmma_f32_16x16x32_bf16(false, a0, false, b,
                                                       (short)0, acc0, false, false);
        acc1 = __builtin_amdgcn_wmma_f32_16x16x32_bf16(false, a1, false, b,
                                                       (short)0, acc1, false, false);
    }

    const int orow = (lane >> 4) << 3;
    if (col < GCN_CLS) {
        float* op0 = t2 + (size_t)(row0 + orow) * GCN_CLS + col;
        float* op1 = t2 + (size_t)(row0 + 16 + orow) * GCN_CLS + col;
#pragma unroll
        for (int r = 0; r < 8; ++r) {
            op0[(size_t)r * GCN_CLS] = acc0[r];
            op1[(size_t)r * GCN_CLS] = acc1[r];
        }
    }
}

// -------------- Edge scatter: agg[dst] += m[src] * w  (atomic f32) ----------
template <int CHUNKS, int F>
__global__ void __launch_bounds__(256)
gcn_scatter(const float* __restrict__ m, const int* __restrict__ src,
            const int* __restrict__ dst, const float* __restrict__ w,
            float* __restrict__ agg, long long total) {
    const long long t = (long long)blockIdx.x * blockDim.x + threadIdx.x;
    if (t >= total) return;
    const int e = (int)(t / CHUNKS);
    const int c = (int)(t % CHUNKS);
    const int s = src[e];
    const int d = dst[e];
    const float wt = w[e];
    const float4 v = ((const float4*)(m + (size_t)s * F))[c];
    float* ap = agg + (size_t)d * F + c * 4;
    unsafeAtomicAdd(ap + 0, v.x * wt);
    unsafeAtomicAdd(ap + 1, v.y * wt);
    unsafeAtomicAdd(ap + 2, v.z * wt);
    unsafeAtomicAdd(ap + 3, v.w * wt);
}

// -------------------- bias + log_softmax over 40 classes --------------------
// One wave32 per row; lane holds elems {lane, lane+32}.
__global__ void __launch_bounds__(256)
gcn_logsoftmax(const float* __restrict__ p, const float* __restrict__ b2,
               float* __restrict__ out, int n) {
    const int lane = threadIdx.x & 31;
    const int wid  = threadIdx.x >> 5;
    const int row  = blockIdx.x * 8 + wid;
    if (row >= n) return;
    const float* pr = p + (size_t)row * GCN_CLS;
    const float x0 = pr[lane] + b2[lane];
    const float x1 = (lane < 8) ? pr[32 + lane] + b2[32 + lane] : -3.0e38f;
    float mx = fmaxf(x0, x1);
#pragma unroll
    for (int off = 16; off > 0; off >>= 1)
        mx = fmaxf(mx, __shfl_xor(mx, off, 32));
    float s = __expf(x0 - mx) + ((lane < 8) ? __expf(x1 - mx) : 0.0f);
#pragma unroll
    for (int off = 16; off > 0; off >>= 1)
        s += __shfl_xor(s, off, 32);
    const float lse = mx + __logf(s);
    float* orow = out + (size_t)row * GCN_CLS;
    orow[lane] = x0 - lse;
    if (lane < 8) orow[32 + lane] = x1 - lse;
}

// ---------------------------------------------------------------------------
extern "C" void kernel_launch(void* const* d_in, const int* in_sizes, int n_in,
                              void* d_out, int out_size, void* d_ws, size_t ws_size,
                              hipStream_t stream) {
    (void)n_in; (void)out_size; (void)ws_size;
    const float* x  = (const float*)d_in[0];
    const int*   ei = (const int*)  d_in[1];
    const float* ew = (const float*)d_in[2];
    const float* W1 = (const float*)d_in[3];
    const float* b1 = (const float*)d_in[4];
    const float* W2 = (const float*)d_in[5];
    const float* b2 = (const float*)d_in[6];
    float* out = (float*)d_out;

    const int E = in_sizes[2];
    const int N = GCN_N;
    const int* src = ei;
    const int* dst = ei + E;

    // Workspace layout (phases reuse buffers):
    //   bufA: m1 (N*128 f32)  then t2 (N*40 f32)
    //   bufB: agg1 (N*128 f32) then agg2 (N*40 f32)
    //   W1t:  128*512 bf16,  W2t: 48*128 bf16
    float*  m1   = (float*)d_ws;
    float*  agg1 = m1 + (size_t)N * GCN_HID;
    float*  t2   = m1;     // gemm1 output dead after scatter1
    float*  agg2 = agg1;   // agg1 dead after gemm2
    __bf16* W1t  = (__bf16*)(agg1 + (size_t)N * GCN_HID);
    __bf16* W2t  = W1t + (size_t)GCN_HID * GCN_INF;

    // Pre-pack weights (bf16, transposed; W2 zero-padded to 48 cols).
    {
        const int total  = GCN_HID * GCN_INF + GCN_CLSP * GCN_HID;
        const int blocks = (total + 255) / 256;
        gcn_pack_weights<<<blocks, 256, 0, stream>>>(W1, W2, W1t, W2t);
    }

    // Layer 1
    hipMemsetAsync(agg1, 0, (size_t)N * GCN_HID * sizeof(float), stream);
    gcn_gemm1_wmma<<<N / G1_ROWS, 128, 0, stream>>>(x, W1t, m1);
    {
        const long long total = (long long)E * (GCN_HID / 4);
        const unsigned blocks = (unsigned)((total + 255) / 256);
        gcn_scatter<GCN_HID / 4, GCN_HID><<<blocks, 256, 0, stream>>>(
            m1, src, dst, ew, agg1, total);
    }

    // Layer 2 (bias + relu fused into gemm2 A staging)
    gcn_gemm2_wmma<<<N / G2_ROWS, 96, 0, stream>>>(agg1, b1, W2t, t2);
    hipMemsetAsync(agg2, 0, (size_t)N * GCN_CLS * sizeof(float), stream);
    {
        const long long total = (long long)E * (GCN_CLS / 4);
        const unsigned blocks = (unsigned)((total + 255) / 256);
        gcn_scatter<GCN_CLS / 4, GCN_CLS><<<blocks, 256, 0, stream>>>(
            t2, src, dst, ew, agg2, total);
    }

    // bias + log_softmax
    gcn_logsoftmax<<<N / 8, 256, 0, stream>>>(agg2, b2, out, N);
}
